// LocalAttention_9526237462630
// MI455X (gfx1250) — compile-verified, matrix operand order checked by reading, not run on hardware
//
#include <hip/hip_runtime.h>
#include <hip/hip_bf16.h>

typedef _Float16 half_t;
typedef __attribute__((ext_vector_type(16))) _Float16 v16h;
typedef __attribute__((ext_vector_type(8)))  float    v8f;
typedef __attribute__((ext_vector_type(2)))  int      v2i;

#define DEV static __device__ __forceinline__

#define GLOBAL_AS __attribute__((address_space(1)))
#define LDS_AS    __attribute__((address_space(3)))

#if defined(__has_builtin)
#if __has_builtin(__builtin_amdgcn_global_load_async_to_lds_b64) && \
    __has_builtin(__builtin_amdgcn_s_wait_asynccnt)
#define USE_ASYNC_LDS 1
#endif
#endif
#ifndef USE_ASYNC_LDS
#define USE_ASYNC_LDS 0
#endif

constexpr int Bb  = 4;
constexpr int T   = 4096;
constexpr int D   = 1024;
constexpr int H   = 16;
constexpr int DH  = 64;
constexpr int WIN = 256;
constexpr int NB  = T / WIN;     // 16 query blocks
constexpr int M   = Bb * T;      // 16384 tokens
constexpr int QKV = 3 * H * DH;  // 3072

// LDS row strides (halves), padded to kill bank conflicts:
// 72 halves = 144B = 36 banks; 40 halves = 80B = 20 banks. Both give 16
// distinct start banks across the 16 fragment lanes (gcd(.,64)=4, period 16).
constexpr int KT_LD = DH + 8;  // 72
constexpr int VT_LD = 32 + 8;  // 40
constexpr int PB_LD = 32 + 8;  // 40

// ---------------------------------------------------------------------------
// WMMA helpers (CDNA5 v_wmma_f32_16x16x32_f16, wave32)
// ---------------------------------------------------------------------------
DEV v8f zero8() {
  v8f z;
#pragma unroll
  for (int i = 0; i < 8; ++i) z[i] = 0.0f;
  return z;
}

DEV v8f wmma_f16(v16h a, v16h b, v8f c) {
  // 8 args: (neg_a, A, neg_b, B, c_mod, C, reuse_a, reuse_b)
  return __builtin_amdgcn_wmma_f32_16x16x32_f16(false, a, false, b, (short)0, c,
                                                false, false);
}

// A/B fragment from a row-major matrix. ISA 16-bit 16x32 layout:
// lane holds row (lane&15); K = g*16 + (lane>>4)*8 + j  (g=0,1; j=0..7).
DEV v16h load_frag_row(const half_t* __restrict__ base, int row_base, int ld,
                       int kbase, int lane) {
  const int row = row_base + (lane & 15);
  const int khi = (lane >> 4) << 3;
  const half_t* p = base + (size_t)row * ld + (kbase + khi);
  v16h f;
#pragma unroll
  for (int j = 0; j < 8; ++j) {
    f[j]     = p[j];
    f[8 + j] = p[16 + j];
  }
  return f;
}

// ---------------------------------------------------------------------------
// Reductions (wave32)
// ---------------------------------------------------------------------------
DEV float wave_reduce_sum(float v) {
#pragma unroll
  for (int m = 16; m >= 1; m >>= 1) v += __shfl_xor(v, m, 32);
  return v;
}
// reduce across the 16 lanes of each half-wave (C-layout rows live per-half)
DEV float half_reduce_max(float v) {
#pragma unroll
  for (int m = 8; m >= 1; m >>= 1) v = fmaxf(v, __shfl_xor(v, m, 32));
  return v;
}
DEV float half_reduce_sum(float v) {
#pragma unroll
  for (int m = 8; m >= 1; m >>= 1) v += __shfl_xor(v, m, 32);
  return v;
}

DEV float block_reduce_sum(float v) {
  __shared__ float sred[16];
  const int lane = threadIdx.x & 31;
  const int wid  = threadIdx.x >> 5;
  __syncthreads();
  v = wave_reduce_sum(v);
  if (lane == 0) sred[wid] = v;
  __syncthreads();
  if (wid == 0) {
    float r = (lane < (int)(blockDim.x >> 5)) ? sred[lane] : 0.0f;
    r = wave_reduce_sum(r);
    if (lane == 0) sred[0] = r;
  }
  __syncthreads();
  float out = sred[0];
  __syncthreads();
  return out;
}

// ---------------------------------------------------------------------------
// Kernel 0: ternary (1.58-bit) weight quantization: w_eff = clip(round(w/s),-1,1)
// stored exactly in f16; per-row scale s = max(mean|w|,1e-5) stored in f32.
// ---------------------------------------------------------------------------
__global__ __launch_bounds__(256) void quant_kernel(
    const float* __restrict__ w, half_t* __restrict__ wq,
    float* __restrict__ sc, int K) {
  const int row = blockIdx.x;
  const float* wr = w + (size_t)row * K;
  float s = 0.0f;
  for (int i = threadIdx.x; i < K; i += blockDim.x) s += fabsf(wr[i]);
  s = block_reduce_sum(s);
  const float scale = fmaxf(s / (float)K, 1e-5f);
  const float inv = 1.0f / scale;
  half_t* qr = wq + (size_t)row * K;
  for (int i = threadIdx.x; i < K; i += blockDim.x) {
    float q = rintf(wr[i] * inv);  // round-half-even, matches jnp.round
    q = fmaxf(-1.0f, fminf(1.0f, q));
    qr[i] = (half_t)q;             // {-1,0,1}: exact in f16
  }
  if (threadIdx.x == 0) sc[row] = scale;
}

// ---------------------------------------------------------------------------
// Kernel 1: LayerNorm (f32 in -> f16 out)
// ---------------------------------------------------------------------------
__global__ __launch_bounds__(256) void ln_kernel(
    const float* __restrict__ x, const float* __restrict__ gw,
    const float* __restrict__ gb, half_t* __restrict__ ho) {
  const int row = blockIdx.x;
  const float* xr = x + (size_t)row * D;
  float s = 0.0f, ss = 0.0f;
  for (int i = threadIdx.x; i < D; i += blockDim.x) {
    const float v = xr[i];
    s += v;
    ss += v * v;
  }
  s  = block_reduce_sum(s);
  ss = block_reduce_sum(ss);
  const float mu   = s / (float)D;
  const float var  = ss / (float)D - mu * mu;
  const float rstd = rsqrtf(var + 1e-5f);
  half_t* hr = ho + (size_t)row * D;
  for (int i = threadIdx.x; i < D; i += blockDim.x)
    hr[i] = (half_t)((xr[i] - mu) * rstd * gw[i] + gb[i]);
}

// ---------------------------------------------------------------------------
// Kernel 2: QKV GEMM  qkv[m,col] = sum_k h[m,k]*wq[col,k], scaled per col.
// Wave computes a 16x64 tile (1 A-frag reused across 4 B-frags per k-chunk).
// Epilogue scatters q,k into [B,H,T,DH] and v TRANSPOSED into [B,H,DH,T]
// (so the attention kernel's P*V B-fragments are row-contiguous).
// q is pre-scaled by DH^-0.5.
// ---------------------------------------------------------------------------
__global__ __launch_bounds__(256) void qkv_gemm(
    const half_t* __restrict__ hm, const half_t* __restrict__ wq,
    const float* __restrict__ sc, half_t* __restrict__ qo,
    half_t* __restrict__ ko, half_t* __restrict__ vo) {
  const int lane = threadIdx.x & 31;
  const int wave = threadIdx.x >> 5;
  const int task = blockIdx.x * 8 + wave;
  const int NTN  = QKV / 64;  // 48
  const int tn = task % NTN;
  const int tm = task / NTN;
  const int mb = tm * 16, nb = tn * 64;

  v8f acc0 = zero8(), acc1 = zero8(), acc2 = zero8(), acc3 = zero8();
  for (int kb = 0; kb < D; kb += 32) {
    __builtin_prefetch(hm + (size_t)mb * D + kb + 128, 0, 3);
    const v16h af = load_frag_row(hm, mb, D, kb, lane);
    const v16h b0 = load_frag_row(wq, nb + 0,  D, kb, lane);
    const v16h b1 = load_frag_row(wq, nb + 16, D, kb, lane);
    const v16h b2 = load_frag_row(wq, nb + 32, D, kb, lane);
    const v16h b3 = load_frag_row(wq, nb + 48, D, kb, lane);
    acc0 = wmma_f16(af, b0, acc0);
    acc1 = wmma_f16(af, b1, acc1);
    acc2 = wmma_f16(af, b2, acc2);
    acc3 = wmma_f16(af, b3, acc3);
  }

  const int g = lane >> 4;
  const v8f accs[4] = {acc0, acc1, acc2, acc3};
#pragma unroll
  for (int t = 0; t < 4; ++t) {
    const int col = nb + t * 16 + (lane & 15);
    float s = sc[col];
    if (col < H * DH) s *= 0.125f;  // q * DH^-0.5
    const int which = col / (H * DH);
    const int cc = col % (H * DH);
    const int hh = cc >> 6;
    const int dd = cc & 63;
    half_t* dst = (which == 0) ? qo : (which == 1) ? ko : vo;
#pragma unroll
    for (int r = 0; r < 8; ++r) {
      const int m    = mb + (g << 3) + r;  // C-layout: M = r + 8*(lane>>4)
      const int bidx = m >> 12;            // / T
      const int tloc = m & (T - 1);
      const size_t addr =
          (which == 2)
              ? (((size_t)bidx * H + hh) * DH + dd) * T + tloc   // v: [B,H,DH,T]
              : (((size_t)bidx * H + hh) * T + tloc) * DH + dd;  // q,k
      dst[addr] = (half_t)(accs[t][r] * s);
    }
  }
}

// ---------------------------------------------------------------------------
// Kernel 3: block-local sliding-window flash attention.
// Grid: B*H*NB blocks of 512 threads (16 waves). Wave w owns queries
// [w*16, w*16+16) of the 256-query block. Key-chunks of 32 keys from the
// 512-key window (prev block + own block) are staged in LDS (K row-major,
// V transposed [dim][key]) and shared by all 16 waves; block 0 skips the
// fully-masked previous-window chunks. Staging uses async global->LDS loads
// (ASYNCcnt) when the builtin is available. Online softmax; P is moved
// C-layout -> A-layout via a per-wave padded LDS buffer.
// ---------------------------------------------------------------------------
__global__ __launch_bounds__(512) void attn_kernel(
    const half_t* __restrict__ qg, const half_t* __restrict__ kg,
    const half_t* __restrict__ vtg, half_t* __restrict__ og) {
  const int blk  = blockIdx.x % NB;
  const int head = (blockIdx.x / NB) % H;
  const int b    = blockIdx.x / (NB * H);
  const int lane = threadIdx.x & 31;
  const int wave = threadIdx.x >> 5;  // 0..15
  const int g    = lane >> 4;

  __shared__ __align__(16) half_t Kt[32][KT_LD];      // keys row-major
  __shared__ __align__(16) half_t Vt[DH][VT_LD];      // V^T: [dim][key]
  __shared__ __align__(16) half_t Pb[16][16][PB_LD];  // per-wave P buffer

  const size_t hoff = ((size_t)(b * H + head)) * T * DH;
  const half_t* Q  = qg + hoff + (size_t)blk * WIN * DH;
  const half_t* K  = kg + hoff;
  const half_t* VT = vtg + hoff;  // [DH][T] for this (b,head)

  // Q fragments: 16 rows x 64 dims = 2 k-chunks of 32 (q already * DH^-0.5)
  const v16h qf0 = load_frag_row(Q, wave * 16, DH, 0, lane);
  const v16h qf1 = load_frag_row(Q, wave * 16, DH, 32, lane);

  float m_i[8], l_i[8];
  v8f oacc[4];
#pragma unroll
  for (int r = 0; r < 8; ++r) { m_i[r] = -1e30f; l_i[r] = 0.0f; }
#pragma unroll
  for (int nd = 0; nd < 4; ++nd) oacc[nd] = zero8();

  const int win0 = blk * WIN - WIN;  // abs token of window position j==0
  // block 0: chunks it<8 are entirely masked (no key j<WIN is valid) -> skip.
  const int it0 = (blk == 0) ? (WIN / 32) : 0;

  for (int it = it0; it < (2 * WIN) / 32; ++it) {
    const int jbase = it * 32;
    const int tok0  = win0 + jbase;  // >= 0 for all processed chunks

    __syncthreads();  // protect previous chunk's LDS reads
    {
      // K staging: 512 threads x 8B  (32 keys x 64 dims)
      const int kk = threadIdx.x >> 4;         // chunk-local key 0..31
      const int cc = (threadIdx.x & 15) << 2;  // dim (4 halves)
      const half_t* gk = K + (size_t)(tok0 + kk) * DH + cc;
      half_t* lk = &Kt[kk][cc];
      // V^T staging: 512 threads x 8B (64 dims x 32 keys)
      const int vd = threadIdx.x >> 3;         // dim 0..63
      const int vk = (threadIdx.x & 7) << 2;   // key (4 halves)
      const half_t* gv = VT + (size_t)vd * T + tok0 + vk;
      half_t* lv = &Vt[vd][vk];
#if USE_ASYNC_LDS
      __builtin_amdgcn_global_load_async_to_lds_b64(
          (GLOBAL_AS v2i*)gk, (LDS_AS v2i*)lk, 0, 0);
      __builtin_amdgcn_global_load_async_to_lds_b64(
          (GLOBAL_AS v2i*)gv, (LDS_AS v2i*)lv, 0, 0);
      __builtin_amdgcn_s_wait_asynccnt(0);
#else
      *(unsigned long long*)lk = *(const unsigned long long*)gk;
      *(unsigned long long*)lv = *(const unsigned long long*)gv;
#endif
    }
    __syncthreads();

    // S = Q (16x64) x K^T (64x32): two 16x16 score tiles
    v8f s0 = zero8(), s1 = zero8();
    {
      const v16h b00 = load_frag_row(&Kt[0][0], 0,  KT_LD, 0,  lane);
      const v16h b01 = load_frag_row(&Kt[0][0], 0,  KT_LD, 32, lane);
      const v16h b10 = load_frag_row(&Kt[0][0], 16, KT_LD, 0,  lane);
      const v16h b11 = load_frag_row(&Kt[0][0], 16, KT_LD, 32, lane);
      s0 = wmma_f16(qf0, b00, s0);
      s0 = wmma_f16(qf1, b01, s0);
      s1 = wmma_f16(qf0, b10, s1);
      s1 = wmma_f16(qf1, b11, s1);
    }

    // mask + online softmax (row = wave*16 + g*8 + r, cols across 16 lanes)
#pragma unroll
    for (int r = 0; r < 8; ++r) {
      const int qrow = wave * 16 + (g << 3) + r;  // query pos within block
      const int j0 = jbase + (lane & 15);
      const int j1 = j0 + 16;
      const int rel0 = j0 - WIN - qrow;
      const int rel1 = j1 - WIN - qrow;
      const bool ok0 =
          (rel0 <= 0) && (rel0 > -WIN) && ((blk > 0) || (j0 >= WIN));
      const bool ok1 =
          (rel1 <= 0) && (rel1 > -WIN) && ((blk > 0) || (j1 >= WIN));
      const float a0 = ok0 ? s0[r] : -3.0e38f;
      const float a1 = ok1 ? s1[r] : -3.0e38f;
      const float mx    = half_reduce_max(fmaxf(a0, a1));
      const float mnew  = fmaxf(m_i[r], mx);
      const float alpha = __expf(m_i[r] - mnew);
      const float p0 = __expf(a0 - mnew);
      const float p1 = __expf(a1 - mnew);
      const float rs = half_reduce_sum(p0 + p1);
      l_i[r] = l_i[r] * alpha + rs;
      m_i[r] = mnew;
#pragma unroll
      for (int nd = 0; nd < 4; ++nd) oacc[nd][r] = oacc[nd][r] * alpha;
      Pb[wave][(g << 3) + r][lane & 15]        = (half_t)p0;
      Pb[wave][(g << 3) + r][16 + (lane & 15)] = (half_t)p1;
    }

    // O += P (16x32) x V (32x64); V B-frag rows are contiguous in Vt
    const v16h pf = load_frag_row(&Pb[wave][0][0], 0, PB_LD, 0, lane);
#pragma unroll
    for (int nd = 0; nd < 4; ++nd) {
      const v16h vf = load_frag_row(&Vt[0][0], nd * 16, VT_LD, 0, lane);
      oacc[nd] = wmma_f16(pf, vf, oacc[nd]);
    }
  }

  // epilogue: O / l -> f16 [B, T, H*DH]
#pragma unroll
  for (int r = 0; r < 8; ++r) {
    const int qrow = wave * 16 + (g << 3) + r;
    const float inv = 1.0f / l_i[r];
    const size_t rowoff =
        ((size_t)b * T + (size_t)blk * WIN + qrow) * (H * DH) + head * DH;
#pragma unroll
    for (int nd = 0; nd < 4; ++nd)
      og[rowoff + nd * 16 + (lane & 15)] = (half_t)(oacc[nd][r] * inv);
  }
}

// ---------------------------------------------------------------------------
// Kernel 4: output projection GEMM (f16 WMMA, f32 out with per-col scale)
// ---------------------------------------------------------------------------
__global__ __launch_bounds__(256) void out_gemm(
    const half_t* __restrict__ am, const half_t* __restrict__ wq,
    const float* __restrict__ sc, float* __restrict__ yo) {
  const int lane = threadIdx.x & 31;
  const int wave = threadIdx.x >> 5;
  const int task = blockIdx.x * 8 + wave;
  const int NTN  = D / 64;  // 16
  const int tn = task % NTN;
  const int tm = task / NTN;
  const int mb = tm * 16, nb = tn * 64;
  const int Kd = H * DH;  // 1024

  v8f acc0 = zero8(), acc1 = zero8(), acc2 = zero8(), acc3 = zero8();
  for (int kb = 0; kb < Kd; kb += 32) {
    __builtin_prefetch(am + (size_t)mb * Kd + kb + 128, 0, 3);
    const v16h af = load_frag_row(am, mb, Kd, kb, lane);
    const v16h b0 = load_frag_row(wq, nb + 0,  Kd, kb, lane);
    const v16h b1 = load_frag_row(wq, nb + 16, Kd, kb, lane);
    const v16h b2 = load_frag_row(wq, nb + 32, Kd, kb, lane);
    const v16h b3 = load_frag_row(wq, nb + 48, Kd, kb, lane);
    acc0 = wmma_f16(af, b0, acc0);
    acc1 = wmma_f16(af, b1, acc1);
    acc2 = wmma_f16(af, b2, acc2);
    acc3 = wmma_f16(af, b3, acc3);
  }

  const int g = lane >> 4;
  const v8f accs[4] = {acc0, acc1, acc2, acc3};
#pragma unroll
  for (int t = 0; t < 4; ++t) {
    const int col = nb + t * 16 + (lane & 15);
    const float s = sc[col];
#pragma unroll
    for (int r = 0; r < 8; ++r) {
      const int m = mb + (g << 3) + r;
      yo[(size_t)m * D + col] = accs[t][r] * s;
    }
  }
}

// ---------------------------------------------------------------------------
// Host-side launcher
// ---------------------------------------------------------------------------
extern "C" void kernel_launch(void* const* d_in, const int* in_sizes, int n_in,
                              void* d_out, int out_size, void* d_ws,
                              size_t ws_size, hipStream_t stream) {
  (void)in_sizes; (void)n_in; (void)out_size; (void)ws_size;
  const float* x      = (const float*)d_in[0];
  const float* norm_w = (const float*)d_in[1];
  const float* norm_b = (const float*)d_in[2];
  const float* qkv_w  = (const float*)d_in[3];  // [3072, 1024]
  const float* out_w  = (const float*)d_in[4];  // [1024, 1024]
  float* out = (float*)d_out;                   // [B, T, D] f32

  char* ws = (char*)d_ws;
  size_t off = 0;
  half_t* hbuf  = (half_t*)(ws + off); off += (size_t)M * D * 2;  // LN out, later attn out
  half_t* qkvwq = (half_t*)(ws + off); off += (size_t)QKV * D * 2;
  half_t* outwq = (half_t*)(ws + off); off += (size_t)D * (H * DH) * 2;
  float* qscale = (float*)(ws + off);  off += (size_t)QKV * 4;
  float* oscale = (float*)(ws + off);  off += (size_t)D * 4;
  half_t* qb  = (half_t*)(ws + off); off += (size_t)Bb * H * T * DH * 2;
  half_t* kb  = (half_t*)(ws + off); off += (size_t)Bb * H * T * DH * 2;
  half_t* vtb = (half_t*)(ws + off); off += (size_t)Bb * H * T * DH * 2;

  quant_kernel<<<QKV, 256, 0, stream>>>(qkv_w, qkvwq, qscale, D);
  quant_kernel<<<D, 256, 0, stream>>>(out_w, outwq, oscale, H * DH);
  ln_kernel<<<M, 256, 0, stream>>>(x, norm_w, norm_b, hbuf);
  qkv_gemm<<<(M / 16) * (QKV / 64) / 8, 256, 0, stream>>>(hbuf, qkvwq, qscale,
                                                          qb, kb, vtb);
  attn_kernel<<<Bb * H * NB, 512, 0, stream>>>(qb, kb, vtb, hbuf);
  out_gemm<<<(M / 16) * (D / 64) / 8, 256, 0, stream>>>(hbuf, outwq, oscale,
                                                        out);
}
